// PointwiseSpAggregatedAttention_47304769798729
// MI455X (gfx1250) — matrix operand order repaired and checked by reading, not compiled, not run on hardware
//
#include <hip/hip_runtime.h>
#include <hip/hip_bf16.h>

typedef __attribute__((ext_vector_type(16))) __bf16 v16bf;
typedef __attribute__((ext_vector_type(8)))  float  v8f;

namespace {
constexpr int S    = 2048;   // sequence length
constexpr int D    = 1024;   // model dim = H * HD
constexpr int H    = 16;     // heads
constexpr int HD   = 64;     // head dim
constexpr int BM   = 128;    // query rows per block (8 waves x 16)
constexpr int BN   = 32;     // key rows per iteration (one bf16 WMMA K-step)
constexpr int NW   = BM / 16;
constexpr int KPAD = 72;     // bf16 row stride for sK  (144 B, conflict-free)
constexpr int VPAD = 40;     // bf16 row stride for sVt (80 B, 16B-aligned, conflict-free)
constexpr int PPAD = 40;     // bf16 row stride for sP
constexpr int BATCH = 2;
constexpr int NKB  = S / BN; // 64 key blocks
}

__global__ __launch_bounds__(256)
void silu_attn_wmma_kernel(const float* __restrict__ Vg,
                           const float* __restrict__ Kg,
                           const float* __restrict__ Qg,
                           float* __restrict__ Og) {
  __shared__ __bf16 sK [2 * BN * KPAD];   // [buf][key][d]
  __shared__ __bf16 sVt[2 * HD * VPAD];   // [buf][d][key]  (transposed V)
  __shared__ __bf16 sP [NW * 16 * PPAD];  // per-wave P staging

  const int tid  = threadIdx.x;
  const int wave = tid >> 5;
  const int lane = tid & 31;
  const int m    = lane & 15;   // row/col index within 16-wide tile
  const int hf   = lane >> 4;   // wave half

  const int qb = blockIdx.x;
  const int bh = blockIdx.y;
  const int b  = bh >> 4;
  const int h  = bh & 15;

  const size_t head_off = (size_t)b * S * D + (size_t)h * HD;
  const float* qh = Qg + head_off;
  const float* kh = Kg + head_off;
  const float* vh = Vg + head_off;
  float*       oh = Og + head_off;

  const int qrow0 = qb * BM + wave * 16;

  // ---- Q: load once, straight into bf16 A-fragment layout (16x32, 2 d-chunks)
  v16bf aQ[2];
#pragma unroll
  for (int c = 0; c < 2; ++c) {
#pragma unroll
    for (int r = 0; r < 8; ++r) {
      const int d = 32 * c + 2 * r + 16 * hf;
      const float2 f = *(const float2*)(qh + (size_t)(qrow0 + m) * D + d);
      aQ[c][2 * r]     = (__bf16)f.x;
      aQ[c][2 * r + 1] = (__bf16)f.y;
    }
  }

  v8f acc[4];
#pragma unroll
  for (int t = 0; t < 4; ++t) acc[t] = {};

  __bf16* sPw = sP + wave * 16 * PPAD;

  // ---- cooperative stage: fp32 global -> bf16 LDS (K row-major, V transposed)
  auto stage = [&](int kb, int buf, int pf_kb) {
    const size_t krow0 = (size_t)kb * BN;
    __bf16* K  = sK  + buf * BN * KPAD;
    __bf16* Vt = sVt + buf * HD * VPAD;
#pragma unroll
    for (int it = 0; it < (BN * HD / 2) / 256; ++it) {
      const int i   = tid + it * 256;   // float2 element index
      const int row = i >> 5;           // 32 float2 per 64-wide row
      const int d2  = i & 31;
      const size_t off = (krow0 + row) * D + 2 * d2;
      const float2 kk = *(const float2*)(kh + off);
      const float2 vv = *(const float2*)(vh + off);
      K [row * KPAD + 2 * d2]       = (__bf16)kk.x;
      K [row * KPAD + 2 * d2 + 1]   = (__bf16)kk.y;
      Vt[(2 * d2) * VPAD + row]     = (__bf16)vv.x;
      Vt[(2 * d2 + 1) * VPAD + row] = (__bf16)vv.y;
      if (pf_kb >= 0) {
        const size_t poff = ((size_t)pf_kb * BN + row) * D + 2 * d2;
        __builtin_prefetch(kh + poff, 0, 0);   // global_prefetch_b8
        __builtin_prefetch(vh + poff, 0, 0);
      }
    }
  };

  stage(0, 0, -1);

  for (int kb = 0; kb < NKB; ++kb) {
    const int cur = kb & 1;
    __syncthreads();                         // staging of 'cur' complete
    if (kb + 1 < NKB)                        // overlap next tile with compute
      stage(kb + 1, cur ^ 1, (kb + 2 < NKB) ? (kb + 2) : -1);

    const __bf16* K  = sK  + cur * BN * KPAD;
    const __bf16* Vt = sVt + cur * HD * VPAD;

    // ---- P[16 x 32] = silu(Q * K^T) for this wave's 16 query rows
#pragma unroll
    for (int t = 0; t < 2; ++t) {
      const int n = t * 16 + m;              // key row (B-operand N index)
      v8f c = {};
#pragma unroll
      for (int ch = 0; ch < 2; ++ch) {       // d chunks of 32
        v16bf bK;
#pragma unroll
        for (int v = 0; v < 8; ++v) {
          const int d = 32 * ch + 2 * v + 16 * hf;
          bK[2 * v]     = K[n * KPAD + d];
          bK[2 * v + 1] = K[n * KPAD + d + 1];
        }
        c = __builtin_amdgcn_wmma_f32_16x16x32_bf16(
                false, aQ[ch], false, bK, (short)0, c, false, false);
      }
      // SiLU: x * rcp(1 + exp(-x)) -- v_exp_f32 + v_rcp_f32, no IEEE div fixup
#pragma unroll
      for (int j = 0; j < 8; ++j) {
        const float x = c[j];
        const float s = x * __builtin_amdgcn_rcpf(1.0f + __expf(-x));
        sPw[(j + 8 * hf) * PPAD + t * 16 + m] = (__bf16)s;
      }
    }

    // ---- reload P as A-fragment (cross-lane reshuffle via LDS)
    v16bf aP;
#pragma unroll
    for (int r = 0; r < 8; ++r) {
      const int k = 2 * r + 16 * hf;
      aP[2 * r]     = sPw[m * PPAD + k];
      aP[2 * r + 1] = sPw[m * PPAD + k + 1];
    }

    // ---- O[16 x 64] += P * V   (bV now 16 contiguous bf16 -> 2x ds_load_b128)
#pragma unroll
    for (int nt = 0; nt < 4; ++nt) {
      const int d = nt * 16 + m;             // output column (B-operand N index)
      v16bf bV;
#pragma unroll
      for (int v = 0; v < 8; ++v) {
        const int kr = 2 * v + 16 * hf;      // consecutive key rows, same column
        bV[2 * v]     = Vt[d * VPAD + kr];
        bV[2 * v + 1] = Vt[d * VPAD + kr + 1];
      }
      acc[nt] = __builtin_amdgcn_wmma_f32_16x16x32_bf16(
                    false, aP, false, bV, (short)0, acc[nt], false, false);
    }
  }

  // ---- epilogue: C/D layout -> global fp32
#pragma unroll
  for (int nt = 0; nt < 4; ++nt) {
#pragma unroll
    for (int j = 0; j < 8; ++j) {
      const int row = qrow0 + j + 8 * hf;
      const int col = nt * 16 + m;
      oh[(size_t)row * D + col] = acc[nt][j];
    }
  }
}

extern "C" void kernel_launch(void* const* d_in, const int* in_sizes, int n_in,
                              void* d_out, int out_size, void* d_ws, size_t ws_size,
                              hipStream_t stream) {
  // setup_inputs dict order: v, k, q
  const float* v = (const float*)d_in[0];
  const float* k = (const float*)d_in[1];
  const float* q = (const float*)d_in[2];
  float* o = (float*)d_out;

  dim3 grid(S / BM, BATCH * H);
  dim3 block(256);
  silu_attn_wmma_kernel<<<grid, block, 0, stream>>>(v, k, q, o);
}
